// FlaxLLaMAAttention_24008867184848
// MI455X (gfx1250) — matrix operand order repaired and checked by reading, not compiled, not run on hardware
//
#include <hip/hip_runtime.h>
#include <hip/hip_bf16.h>

typedef __attribute__((ext_vector_type(16))) __bf16 v16bf;
typedef __attribute__((ext_vector_type(8)))  __bf16 v8bf;
typedef __attribute__((ext_vector_type(4)))  __bf16 v4bf;
typedef __attribute__((ext_vector_type(8)))  float  v8f;

#define WMMA_BF16(a, b, c) \
  __builtin_amdgcn_wmma_f32_16x16x32_bf16(false, (a), false, (b), (short)0, (c), false, false)

#define CAT16(lo, hi) __builtin_shufflevector((lo), (hi), 0,1,2,3,4,5,6,7,8,9,10,11,12,13,14,15)

// CDNA5 async global->LDS copy (16B). GV mode: 64-bit vaddr, saddr=off.
// LDS destination address = low 32 bits of the flat pointer (ISA 10.2).
__device__ __forceinline__ void async_ld16(const __bf16* g, __bf16* l) {
  asm volatile("global_load_async_to_lds_b128 %0, %1, off"
               :: "v"((unsigned)(unsigned long long)l),
                  "v"((unsigned long long)g)
               : "memory");
}
#define WAIT_ASYNC(n) asm volatile("s_wait_asynccnt " #n ::: "memory")

// ---------------------------------------------------------------------------
// f32 -> bf16 elementwise convert (hidden states), 4 elems/thread
// ---------------------------------------------------------------------------
__global__ void cvt_f32_bf16(const float* __restrict__ in, __bf16* __restrict__ out) {
  size_t i = ((size_t)blockIdx.x * 256 + threadIdx.x) * 4;
  const float4 v = *reinterpret_cast<const float4*>(in + i);
  v4bf o;
  o[0] = (__bf16)v.x; o[1] = (__bf16)v.y; o[2] = (__bf16)v.z; o[3] = (__bf16)v.w;
  *reinterpret_cast<v4bf*>(out + i) = o;
}

// ---------------------------------------------------------------------------
// Weight transpose: W f32 [K=2048][N=2048] -> Wt bf16 [N][K]
// ---------------------------------------------------------------------------
__global__ void wtrans_bf16(const float* __restrict__ W, __bf16* __restrict__ Wt) {
  __shared__ float tile[32][33];
  const int tx = threadIdx.x, ty = threadIdx.y;
  const int n0 = blockIdx.x * 32, k0 = blockIdx.y * 32;
#pragma unroll
  for (int i = 0; i < 4; ++i)
    tile[ty + i * 8][tx] = W[(size_t)(k0 + ty + i * 8) * 2048 + n0 + tx];
  __syncthreads();
#pragma unroll
  for (int i = 0; i < 4; ++i)
    Wt[(size_t)(n0 + ty + i * 8) * 2048 + k0 + tx] = (__bf16)tile[tx][ty + i * 8];
}

// ---------------------------------------------------------------------------
// Tiled bf16 GEMM: C[M=4096][N=2048] = A[M][K=2048] * Bt[N][K]^T
// 128x128 block tile, 256 threads (8 waves, 2x4 WMMA tiles each), K-step 32.
// Tiles staged via async global->LDS with double buffering + partial waits.
// EPI: 0 = RoPE + bf16 -> [B,H,S,hd]      (Q/K projections)
//      1 = bf16 transposed -> [B,H,hd,S]  (V projection)
//      2 = f32 row-major -> d_out         (output projection)
// ---------------------------------------------------------------------------
template <int EPI>
__global__ __launch_bounds__(256) void gemm_bf16(const __bf16* __restrict__ A,
                                                 const __bf16* __restrict__ Bt,
                                                 void* __restrict__ outp) {
  constexpr int Kd = 2048, NK = Kd / 32;
  constexpr int LDP = 40;  // 32 + 8 pad: 16B-aligned rows, conflict-free banks
  __shared__ __align__(16) __bf16 As[2][128][LDP];
  __shared__ __align__(16) __bf16 Bs[2][128][LDP];

  const int t = threadIdx.x;
  const int lane = t & 31, wid = t >> 5;
  const int wm = wid >> 1, wn = wid & 1;
  const int nl = lane & 15, hig = lane >> 4;
  const int bM = blockIdx.y * 128, bN = blockIdx.x * 128;

  v8f acc[2][4];
#pragma unroll
  for (int mt = 0; mt < 2; ++mt)
#pragma unroll
    for (int nt = 0; nt < 4; ++nt)
#pragma unroll
      for (int r = 0; r < 8; ++r) acc[mt][nt][r] = 0.0f;

  // Each thread owns 32B of A and 32B of Bt per K-tile.
  const int lr = t >> 1;
  const int lc = (t & 1) * 16;
  const __bf16* ga = A  + (size_t)(bM + lr) * Kd + lc;
  const __bf16* gb = Bt + (size_t)(bN + lr) * Kd + lc;

  // Prologue: async-stage tile 0 into buffer 0.
  async_ld16(ga,     &As[0][lr][lc]);
  async_ld16(ga + 8, &As[0][lr][lc + 8]);
  async_ld16(gb,     &Bs[0][lr][lc]);
  async_ld16(gb + 8, &Bs[0][lr][lc + 8]);

  for (int kt = 0; kt < NK; ++kt) {
    const int cur = kt & 1;
    if (kt + 1 < NK) {
      const __bf16* na = ga + (size_t)(kt + 1) * 32;
      const __bf16* nb = gb + (size_t)(kt + 1) * 32;
      const int nxt = cur ^ 1;
      async_ld16(na,     &As[nxt][lr][lc]);
      async_ld16(na + 8, &As[nxt][lr][lc + 8]);
      async_ld16(nb,     &Bs[nxt][lr][lc]);
      async_ld16(nb + 8, &Bs[nxt][lr][lc + 8]);
      if (kt + 2 < NK) {  // warm L2 for the tile after next
        __builtin_prefetch(ga + (size_t)(kt + 2) * 32, 0, 0);
        __builtin_prefetch(gb + (size_t)(kt + 2) * 32, 0, 0);
      }
      WAIT_ASYNC(4);  // retire current tile's 4 loads; keep next 4 in flight
    } else {
      WAIT_ASYNC(0);
    }
    __syncthreads();

    // A fragment: lane row m=nl, K = {c0..c0+7, c0+16..c0+23}, c0 = 8*hig
    v16bf af[2];
#pragma unroll
    for (int mt = 0; mt < 2; ++mt) {
      const int row = wm * 32 + mt * 16 + nl;
      const int c0 = hig * 8;
      v8bf lo = *(const v8bf*)&As[cur][row][c0];
      v8bf hh = *(const v8bf*)&As[cur][row][c0 + 16];
      af[mt] = CAT16(lo, hh);
    }
    // B fragment: lane column n=nl, K = {k0..k0+7, k0+16..k0+23}, k0 = 8*hig
    v16bf bfm[4];
#pragma unroll
    for (int nt = 0; nt < 4; ++nt) {
      const int nr = wn * 64 + nt * 16 + nl;
      const int k0 = hig * 8;
      v8bf lo = *(const v8bf*)&Bs[cur][nr][k0];
      v8bf hh = *(const v8bf*)&Bs[cur][nr][k0 + 16];
      bfm[nt] = CAT16(lo, hh);
    }
#pragma unroll
    for (int mt = 0; mt < 2; ++mt)
#pragma unroll
      for (int nt = 0; nt < 4; ++nt)
        acc[mt][nt] = WMMA_BF16(af[mt], bfm[nt], acc[mt][nt]);
    __syncthreads();  // everyone done with buffer `cur` before re-staging it
  }

  // Epilogue. C layout: lane column n = nl; rows m = 8*hig + r (r = VGPR idx).
#pragma unroll
  for (int mt = 0; mt < 2; ++mt) {
#pragma unroll
    for (int nt = 0; nt < 4; ++nt) {
      const int col = bN + wn * 64 + nt * 16 + nl;
      const int rowbase = bM + wm * 32 + mt * 16 + hig * 8;
      if (EPI == 2) {
        float* out = (float*)outp;
#pragma unroll
        for (int r = 0; r < 8; ++r)
          out[(size_t)(rowbase + r) * 2048 + col] = acc[mt][nt][r];
      } else if (EPI == 1) {
        __bf16* out = (__bf16*)outp;
        const int h = col >> 7, d = col & 127;
#pragma unroll
        for (int r = 0; r < 8; ++r) {
          const int row = rowbase + r;
          const int spos = row & 2047, b = row >> 11;
          out[((size_t)(b * 16 + h) * 128 + d) * 2048 + spos] = (__bf16)acc[mt][nt][r];
        }
      } else {  // EPI == 0: RoPE fused
        __bf16* out = (__bf16*)outp;
        const int h = col >> 7, d = col & 127;
        // inv_freq = 10000^(-2*(d/2)/128) = exp(-(d/2)*ln(1e4)/64)
        const float invf = __expf(-0.14391157f * (float)(d >> 1));
#pragma unroll
        for (int r = 0; r < 8; ++r) {
          const int row = rowbase + r;
          const int spos = row & 2047, b = row >> 11;
          float sn, cs;
          __sincosf((float)spos * invf, &sn, &cs);
          const float v  = acc[mt][nt][r];
          const float vp = __shfl_xor(v, 1, 32);  // partner head-dim (d^1)
          const float res = (d & 1) ? (vp * sn + v * cs) : (v * cs - vp * sn);
          out[((size_t)(b * 16 + h) * 2048 + spos) * 128 + d] = (__bf16)res;
        }
      }
    }
  }
}

// ---------------------------------------------------------------------------
// Flash attention: 4 waves / block, 64 query rows (16 per wave), one (b,h).
// Each 32-key chunk of K and V is async-staged into LDS ONCE and consumed by
// all 4 waves (4x less L2 traffic). Online softmax per wave.
// Q,K: bf16 [B*H][S][128] (RoPE applied). Vt: bf16 [B*H][128][S].
// O:  bf16 [B][S][2048] (ready as A-matrix for the output projection).
// ---------------------------------------------------------------------------
__global__ __launch_bounds__(128) void flash_attn(const __bf16* __restrict__ Q,
                                                  const __bf16* __restrict__ K,
                                                  const __bf16* __restrict__ Vt,
                                                  __bf16* __restrict__ O) {
  constexpr int S = 2048, HD = 128;
  __shared__ __align__(16) __bf16 Ks[2][32][136];  // [key][d], 272B rows
  __shared__ __align__(16) __bf16 Vs[2][128][40];  // [d][key], 80B rows
  __shared__ __align__(16) __bf16 Plds[4][16][40]; // per-wave P tile

  const int t = threadIdx.x;
  const int lane = t & 31, w = t >> 5;
  const int nl = lane & 15, hig = lane >> 4;
  const int q0b = blockIdx.x * 64;        // block's first query row
  const int wq0 = q0b + w * 16;           // this wave's first query row
  const int bh = blockIdx.y;

  const __bf16* Qp = Q  + (size_t)bh * S * HD;
  const __bf16* Kp = K  + (size_t)bh * S * HD;
  const __bf16* Vp = Vt + (size_t)bh * HD * S;

  // Q fragments (resident): lane row m=nl of this wave's 16 rows.
  v16bf qf[4];
  {
    const __bf16* qbase = Qp + (size_t)(wq0 + nl) * HD;
#pragma unroll
    for (int ch = 0; ch < 4; ++ch) {
      const int c0 = ch * 32 + hig * 8;
      v8bf lo = *(const v8bf*)(qbase + c0);
      v8bf hh = *(const v8bf*)(qbase + c0 + 16);
      qf[ch] = CAT16(lo, hh);
    }
  }

  v8f o[8];
#pragma unroll
  for (int n = 0; n < 8; ++n)
#pragma unroll
    for (int r = 0; r < 8; ++r) o[n][r] = 0.0f;
  float m_i[8], l_i[8];
#pragma unroll
  for (int r = 0; r < 8; ++r) { m_i[r] = -3.0e38f; l_i[r] = 0.0f; }

  const float SCALE = 0.08838834764831845f;  // 1/sqrt(128)
  const int jmax = q0b + 63;                 // block-diagonal bound (< S always)

  // Cooperative async stage of one 32-key chunk: K 8KB + V columns 8KB.
  const int kr = t >> 2, cseg = (t & 3) * 32;  // K: row kr (of 32), 64B segment
  auto issue = [&](int j0, int buf) {
    const __bf16* gk = Kp + (size_t)(j0 + kr) * HD + cseg;
    __bf16* lk = &Ks[buf][kr][cseg];
#pragma unroll
    for (int i = 0; i < 4; ++i) async_ld16(gk + i * 8, lk + i * 8);
    const __bf16* gv = Vp + (size_t)t * S + j0;  // V: d-row t, 32 keys = 64B
    __bf16* lv = &Vs[buf][t][0];
#pragma unroll
    for (int i = 0; i < 4; ++i) async_ld16(gv + i * 8, lv + i * 8);
  };

  issue(0, 0);
  for (int j0 = 0; j0 <= jmax; j0 += 32) {
    const int cur = (j0 >> 5) & 1;
    if (j0 + 32 <= jmax) {
      issue(j0 + 32, cur ^ 1);
      WAIT_ASYNC(8);  // retire current chunk's 8 loads, keep next 8 in flight
    } else {
      WAIT_ASYNC(0);
    }
    __syncthreads();

    if (j0 <= wq0 + 15) {  // wave-uniform causal skip
      // --- S = Q . K^T (two 16x16 tiles: keys j0..+15 and j0+16..+31) ---
      v8f sv[2];
#pragma unroll
      for (int jh = 0; jh < 2; ++jh) {
        v8f c = {0.f, 0.f, 0.f, 0.f, 0.f, 0.f, 0.f, 0.f};
#pragma unroll
        for (int ch = 0; ch < 4; ++ch) {
          const int d0 = ch * 32 + hig * 8;
          v8bf lo = *(const v8bf*)&Ks[cur][jh * 16 + nl][d0];
          v8bf hh = *(const v8bf*)&Ks[cur][jh * 16 + nl][d0 + 16];
          v16bf kf = CAT16(lo, hh);
          c = WMMA_BF16(qf[ch], kf, c);
        }
        sv[jh] = c;
      }

      // --- scale + causal mask + rowwise max over 16-lane group ---
      float mrow[8];
#pragma unroll
      for (int r = 0; r < 8; ++r) {
        const int qidx = wq0 + hig * 8 + r;
#pragma unroll
        for (int jh = 0; jh < 2; ++jh) {
          const int kidx = j0 + jh * 16 + nl;
          float v = sv[jh][r] * SCALE;
          if (kidx > qidx) v = -3.0e38f;
          sv[jh][r] = v;
        }
        float v = fmaxf(sv[0][r], sv[1][r]);
        v = fmaxf(v, __shfl_xor(v, 1, 32));
        v = fmaxf(v, __shfl_xor(v, 2, 32));
        v = fmaxf(v, __shfl_xor(v, 4, 32));
        v = fmaxf(v, __shfl_xor(v, 8, 32));
        mrow[r] = v;
      }

      // --- online softmax; write P (bf16) to this wave's LDS tile ---
#pragma unroll
      for (int r = 0; r < 8; ++r) {
        const float mnew = fmaxf(m_i[r], mrow[r]);
        const float sc = __expf(m_i[r] - mnew);
        m_i[r] = mnew;
        const float p0 = __expf(sv[0][r] - mnew);
        const float p1 = __expf(sv[1][r] - mnew);
        float rs = p0 + p1;
        rs += __shfl_xor(rs, 1, 32);
        rs += __shfl_xor(rs, 2, 32);
        rs += __shfl_xor(rs, 4, 32);
        rs += __shfl_xor(rs, 8, 32);
        l_i[r] = l_i[r] * sc + rs;
#pragma unroll
        for (int n = 0; n < 8; ++n) o[n][r] *= sc;
        Plds[w][hig * 8 + r][nl]      = (__bf16)p0;
        Plds[w][hig * 8 + r][16 + nl] = (__bf16)p1;
      }
      asm volatile("s_wait_dscnt 0" ::: "memory");  // intra-wave P RAW fence

      // --- P fragment (A layout) ---
      v16bf pf;
      {
        const int c0 = hig * 8;
        v8bf lo = *(const v8bf*)&Plds[w][nl][c0];
        v8bf hh = *(const v8bf*)&Plds[w][nl][c0 + 16];
        pf = CAT16(lo, hh);
      }

      // --- O += P . V (8 output chunks of 16 dims, K-dim = 32 keys) ---
#pragma unroll
      for (int n = 0; n < 8; ++n) {
        v8bf lo = *(const v8bf*)&Vs[cur][n * 16 + nl][hig * 8];
        v8bf hh = *(const v8bf*)&Vs[cur][n * 16 + nl][hig * 8 + 16];
        v16bf vf = CAT16(lo, hh);
        o[n] = WMMA_BF16(pf, vf, o[n]);
      }
    }
    __syncthreads();  // all waves done with buffer `cur` before re-staging
  }

  // --- finalize: O /= l, write bf16 [B][S][H*hd] ---
  const int b = bh >> 4, h = bh & 15;
#pragma unroll
  for (int r = 0; r < 8; ++r) {
    const float inv = 1.0f / l_i[r];
    const int row = wq0 + hig * 8 + r;
    __bf16* orow = O + ((size_t)(b * 2048 + row) * 2048) + h * 128;
#pragma unroll
    for (int n = 0; n < 8; ++n)
      orow[n * 16 + nl] = (__bf16)(o[n][r] * inv);
  }
}

// ---------------------------------------------------------------------------
extern "C" void kernel_launch(void* const* d_in, const int* in_sizes, int n_in,
                              void* d_out, int out_size, void* d_ws, size_t ws_size,
                              hipStream_t stream) {
  (void)in_sizes; (void)n_in; (void)out_size; (void)ws_size;
  const float* hid = (const float*)d_in[0];
  const float* wq  = (const float*)d_in[1];
  const float* wk  = (const float*)d_in[2];
  const float* wv  = (const float*)d_in[3];
  const float* wo  = (const float*)d_in[4];

  char* ws = (char*)d_ws;
  const size_t MB = 1ull << 20;
  __bf16* Xb  = (__bf16*)(ws + 0);        // 16 MB: hidden bf16 [4096][2048]
  __bf16* Wqt = (__bf16*)(ws + 16 * MB);  //  8 MB each: W^T bf16 [N][K]
  __bf16* Wkt = (__bf16*)(ws + 24 * MB);
  __bf16* Wvt = (__bf16*)(ws + 32 * MB);
  __bf16* Wot = (__bf16*)(ws + 40 * MB);
  __bf16* Qb  = (__bf16*)(ws + 48 * MB);  // 16 MB: [B,H,S,hd] (RoPE'd)
  __bf16* Kb  = (__bf16*)(ws + 64 * MB);  // 16 MB: [B,H,S,hd] (RoPE'd)
  __bf16* Vtb = (__bf16*)(ws + 80 * MB);  // 16 MB: [B,H,hd,S]
  __bf16* Ob  = (__bf16*)(ws + 96 * MB);  // 16 MB: attention out [4096][2048]

  cvt_f32_bf16<<<8192, 256, 0, stream>>>(hid, Xb);
  dim3 tb(32, 8), tg(64, 64);
  wtrans_bf16<<<tg, tb, 0, stream>>>(wq, Wqt);
  wtrans_bf16<<<tg, tb, 0, stream>>>(wk, Wkt);
  wtrans_bf16<<<tg, tb, 0, stream>>>(wv, Wvt);
  wtrans_bf16<<<tg, tb, 0, stream>>>(wo, Wot);

  dim3 gg(16, 32);  // N/128 x M/128
  gemm_bf16<0><<<gg, 256, 0, stream>>>(Xb, Wqt, Qb);
  gemm_bf16<0><<<gg, 256, 0, stream>>>(Xb, Wkt, Kb);
  gemm_bf16<1><<<gg, 256, 0, stream>>>(Xb, Wvt, Vtb);

  flash_attn<<<dim3(32, 32), 128, 0, stream>>>(Qb, Kb, Vtb, Ob);

  gemm_bf16<2><<<gg, 256, 0, stream>>>(Ob, Wot, d_out);
}
